// AttentionModuleTriModal_36644660969674
// MI455X (gfx1250) — matrix operand order repaired and checked by reading, not compiled, not run on hardware
//
#include <hip/hip_runtime.h>
#include <hip/hip_bf16.h>

// ---------------------------------------------------------------------------
// AttentionModuleTriModal for MI455X (gfx1250, wave32, WMMA).
// Strategy: convert all fp32 operands to bf16 once, run every GEMM through
// V_WMMA_F32_16X16X32_BF16 with fp32 accumulation (compute-bound problem:
// ~0.86 TFLOP vs ~1 GB of HBM traffic), softmax in fp32.
// Tile staging uses CDNA5 async global->LDS DMA (ASYNCcnt) where the layout
// allows, bypassing the VGPR round trip.
// ---------------------------------------------------------------------------

typedef __attribute__((ext_vector_type(16))) __bf16 bf16x16;
typedef __attribute__((ext_vector_type(8)))  __bf16 bf16x8;
typedef __attribute__((ext_vector_type(8)))  float  f32x8;

#define NN 16384
#define DD 2048

static constexpr int BM = 128;
static constexpr int BN = 128;
static constexpr int BK = 32;
static constexpr int LDP = BK + 8;   // LDS row padding (keeps 16B alignment, skews banks)

// ---------------------------------------------------------------------------
// CDNA5 async global->LDS copy, 16 bytes per lane. LDS flat addresses carry
// the LDS byte offset in the low 32 bits, so truncating the generic pointer
// yields the VDST operand the instruction wants.
// ---------------------------------------------------------------------------
__device__ __forceinline__ void async_copy_b128(const __bf16* g, __bf16* l) {
  unsigned int       lds  = (unsigned int)(unsigned long long)(uintptr_t)l;
  unsigned long long gadr = (unsigned long long)(uintptr_t)g;
  asm volatile("global_load_async_to_lds_b128 %0, %1, off"
               :: "v"(lds), "v"(gadr) : "memory");
}
__device__ __forceinline__ void wait_async0() {
  asm volatile("s_wait_asynccnt 0x0" ::: "memory");
}

// Branch-free tanh: 1 - 2/(e^{2x}+1). Correct limits at +/-inf.
__device__ __forceinline__ float fast_tanh(float x) {
  float e = __expf(2.0f * x);
  return 1.0f - 2.0f * __builtin_amdgcn_rcpf(e + 1.0f);
}

// ---------------------------------------------------------------------------
// f32 -> bf16 conversion (vectorized x4). n must be a multiple of 4.
// ---------------------------------------------------------------------------
__global__ void cvt_f32_to_bf16(const float* __restrict__ in,
                                __bf16* __restrict__ out, long n) {
  long i = ((long)blockIdx.x * blockDim.x + threadIdx.x) * 4;
  if (i + 3 < n) {
    float4 f = *reinterpret_cast<const float4*>(in + i);
    out[i + 0] = (__bf16)f.x;
    out[i + 1] = (__bf16)f.y;
    out[i + 2] = (__bf16)f.z;
    out[i + 3] = (__bf16)f.w;
  }
}

// ---------------------------------------------------------------------------
// Tiled WMMA GEMM.
//   C[M, Ncol] = op(A[M,K] * B) (+ bias[col]) (tanh)
//   TRANS_B=true : B is [Ncol, K] row-major  (computes A @ B^T, dot-of-rows)
//   TRANS_B=false: B is [K, Ncol] row-major  (computes A @ B)
// All dims assumed multiples of 128 (true for N=16384, D=2048).
// Block = 256 threads = 8 waves; tile 128x128x32; each wave owns 32x64
// (2x4 WMMA accumulators). Double-buffered LDS, async-DMA staging.
// ---------------------------------------------------------------------------
template <bool TRANS_B, bool OUT_BF16, bool DO_TANH, bool DO_BIAS>
__global__ void __launch_bounds__(256)
gemm_wmma_bf16(const __bf16* __restrict__ A, const __bf16* __restrict__ B,
               const float* __restrict__ bias,
               float* __restrict__ outF, __bf16* __restrict__ outB,
               int M, int Ncol, int K) {
  __shared__ __bf16 As[2][BM][LDP];
  __shared__ __bf16 Bs[2][BN][LDP];

  const int tid   = threadIdx.x;
  const int lane  = tid & 31;
  const int wave  = tid >> 5;
  const int waveM = wave >> 1;   // 0..3 -> 32-row strip
  const int waveN = wave & 1;    // 0..1 -> 64-col strip
  const int mBase = blockIdx.y * BM;
  const int nBase = blockIdx.x * BN;
  const int l16   = lane & 15;
  const int lHi   = lane >> 4;   // half-wave select

  f32x8 acc[2][4];
#pragma unroll
  for (int i = 0; i < 2; ++i)
#pragma unroll
    for (int j = 0; j < 4; ++j)
#pragma unroll
      for (int e = 0; e < 8; ++e) acc[i][j][e] = 0.0f;

  auto stage = [&](int kt, int buf) {
    const int k0 = kt * BK;
    // A tile: 128 rows x 32 bf16 = 512 16B-chunks, 2 per thread, async DMA.
#pragma unroll
    for (int i = 0; i < 2; ++i) {
      int c   = tid + i * 256;
      int row = c >> 2;
      int cir = c & 3;
      async_copy_b128(A + (size_t)(mBase + row) * K + k0 + cir * 8,
                      &As[buf][row][cir * 8]);
    }
    if constexpr (TRANS_B) {
      // B rows are K-major already: async-stage row n as Bs[n][k].
#pragma unroll
      for (int i = 0; i < 2; ++i) {
        int c   = tid + i * 256;
        int row = c >> 2;
        int cir = c & 3;
        async_copy_b128(B + (size_t)(nBase + row) * K + k0 + cir * 8,
                        &Bs[buf][row][cir * 8]);
      }
    } else {
      // B is [K, Ncol]: transpose while staging so Bs[n][k] is K-contiguous.
#pragma unroll
      for (int i = 0; i < 2; ++i) {
        int c  = tid + i * 256;   // 512 chunks of 8 cols
        int k  = c >> 4;          // 0..31
        int nc = c & 15;          // 0..15
        bf16x8 d = *reinterpret_cast<const bf16x8*>(
            B + (size_t)(k0 + k) * Ncol + nBase + nc * 8);
#pragma unroll
        for (int j = 0; j < 8; ++j) Bs[buf][nc * 8 + j][k] = d[j];
      }
    }
  };

  stage(0, 0);
  wait_async0();
  __syncthreads();

  const int KT = K / BK;
  for (int kt = 0; kt < KT; ++kt) {
    const int buf = kt & 1;
    if (kt + 1 < KT) stage(kt + 1, buf ^ 1);   // DMA overlaps the WMMAs below

    // A fragments: lane L(0..15) holds row M=L, K {0..7,16..23};
    //              lane L+16 holds row M=L, K {8..15,24..31}.
    bf16x16 afrag[2];
    const int koffA = lHi * 8;
#pragma unroll
    for (int i = 0; i < 2; ++i) {
      const int r = waveM * 32 + i * 16 + l16;
      bf16x8 lo = *reinterpret_cast<const bf16x8*>(&As[buf][r][koffA]);
      bf16x8 hi = *reinterpret_cast<const bf16x8*>(&As[buf][r][koffA + 16]);
#pragma unroll
      for (int e = 0; e < 8; ++e) { afrag[i][e] = lo[e]; afrag[i][8 + e] = hi[e]; }
    }
    // B fragments: lane L(0..15) col N=L holds K {0..7,16..23}; lane L+16 K {8..15,24..31}.
    bf16x16 bfrag[4];
    const int kbB = lHi * 8;
#pragma unroll
    for (int j = 0; j < 4; ++j) {
      const int cidx = waveN * 64 + j * 16 + l16;
      bf16x8 lo = *reinterpret_cast<const bf16x8*>(&Bs[buf][cidx][kbB]);
      bf16x8 hi = *reinterpret_cast<const bf16x8*>(&Bs[buf][cidx][kbB + 16]);
#pragma unroll
      for (int e = 0; e < 8; ++e) { bfrag[j][e] = lo[e]; bfrag[j][8 + e] = hi[e]; }
    }
#pragma unroll
    for (int i = 0; i < 2; ++i)
#pragma unroll
      for (int j = 0; j < 4; ++j)
        acc[i][j] = __builtin_amdgcn_wmma_f32_16x16x32_bf16(
            false, afrag[i], false, bfrag[j], (short)0, acc[i][j], false, false);

    wait_async0();   // own async writes committed to LDS
    __syncthreads(); // everyone's committed -> next buffer readable
  }

  // Epilogue. C/D layout: VGPR e -> M = e (+8 for upper half-wave), N = lane&15.
#pragma unroll
  for (int i = 0; i < 2; ++i) {
#pragma unroll
    for (int j = 0; j < 4; ++j) {
      const int rowBase = mBase + waveM * 32 + i * 16 + lHi * 8;
      const int col     = nBase + waveN * 64 + j * 16 + l16;
      float bv = 0.0f;
      if constexpr (DO_BIAS) bv = bias[col];
#pragma unroll
      for (int e = 0; e < 8; ++e) {
        float v = acc[i][j][e];
        if constexpr (DO_BIAS) v += bv;
        if constexpr (DO_TANH) v = fast_tanh(v);
        const size_t idx = (size_t)(rowBase + e) * Ncol + col;
        if constexpr (OUT_BF16) outB[idx] = (__bf16)v;
        else                    outF[idx] = v;
      }
    }
  }
}

// ---------------------------------------------------------------------------
// Row softmax over [rows, L] fp32 -> bf16. One 256-thread block per row,
// L/256 = 64 values per thread held in registers; LDS tree reductions.
// ---------------------------------------------------------------------------
__global__ void __launch_bounds__(256)
softmax_rows_f32_to_bf16(const float* __restrict__ in,
                         __bf16* __restrict__ out, int L) {
  const int row = blockIdx.x;
  const float* r = in + (size_t)row * L;
  __shared__ float red[256];

  float v[64];
  float m = -3.402823466e38f;
#pragma unroll
  for (int i = 0; i < 64; ++i) {
    v[i] = r[threadIdx.x + i * 256];
    m = fmaxf(m, v[i]);
  }
  red[threadIdx.x] = m;
  __syncthreads();
  for (int s = 128; s > 0; s >>= 1) {
    if (threadIdx.x < s) red[threadIdx.x] = fmaxf(red[threadIdx.x], red[threadIdx.x + s]);
    __syncthreads();
  }
  m = red[0];
  __syncthreads();

  float sum = 0.0f;
#pragma unroll
  for (int i = 0; i < 64; ++i) {
    v[i] = __expf(v[i] - m);
    sum += v[i];
  }
  red[threadIdx.x] = sum;
  __syncthreads();
  for (int s = 128; s > 0; s >>= 1) {
    if (threadIdx.x < s) red[threadIdx.x] += red[threadIdx.x + s];
    __syncthreads();
  }
  const float inv = 1.0f / red[0];

  __bf16* o = out + (size_t)row * L;
#pragma unroll
  for (int i = 0; i < 64; ++i)
    o[threadIdx.x + i * 256] = (__bf16)(v[i] * inv);
}

// ---------------------------------------------------------------------------
// bf16 transpose: in[R,C] -> out[C,R]. 32x32 tiles, 256 threads.
// ---------------------------------------------------------------------------
__global__ void __launch_bounds__(256)
transpose_bf16(const __bf16* __restrict__ in, __bf16* __restrict__ out,
               int R, int C) {
  __shared__ __bf16 tile[32][33];
  const int tx = threadIdx.x & 31;
  const int ty = threadIdx.x >> 5;  // 0..7
  const int x  = blockIdx.x * 32 + tx;
#pragma unroll
  for (int j = 0; j < 32; j += 8)
    tile[ty + j][tx] = in[(size_t)(blockIdx.y * 32 + ty + j) * C + x];
  __syncthreads();
  const int x2 = blockIdx.y * 32 + tx;
#pragma unroll
  for (int j = 0; j < 32; j += 8)
    out[(size_t)(blockIdx.x * 32 + ty + j) * R + x2] = tile[tx][ty + j];
}

// ---------------------------------------------------------------------------
extern "C" void kernel_launch(void* const* d_in, const int* in_sizes, int n_in,
                              void* d_out, int out_size, void* d_ws, size_t ws_size,
                              hipStream_t stream) {
  const float* x1 = (const float*)d_in[0];
  const float* x2 = (const float*)d_in[1];
  const float* x3 = (const float*)d_in[2];
  const float* W1 = (const float*)d_in[3];
  const float* b1 = (const float*)d_in[4];
  const float* W2 = (const float*)d_in[5];
  const float* b2 = (const float*)d_in[6];

  char* ws = (char*)d_ws;
  size_t off = 0;
  auto take = [&](size_t bytes) -> char* {
    char* p = ws + off;
    off += (bytes + 255) & ~(size_t)255;
    return p;
  };

  const size_t ND  = (size_t)NN * DD;
  const size_t DDn = (size_t)DD * DD;

  __bf16* x1b = (__bf16*)take(ND * 2);
  __bf16* x2b = (__bf16*)take(DDn * 2);
  __bf16* x3b = (__bf16*)take(DDn * 2);
  __bf16* W1b = (__bf16*)take(DDn * 2);
  __bf16* W2b = (__bf16*)take(DDn * 2);
  __bf16* h1b = (__bf16*)take(ND * 2);    // later reused as weights [D,N]
  __bf16* p2b = (__bf16*)take(DDn * 2);
  __bf16* h3b = (__bf16*)take(DDn * 2);
  __bf16* scb = (__bf16*)take(ND * 2);    // scores [N,D] bf16
  float*  s2f = (float*)take(ND * 4);     // scores2 [D,N] f32; later reused as weightsT

  __bf16* weights  = h1b;                 // [D, N] (h1 dead by softmax time)
  __bf16* weightsT = (__bf16*)s2f;        // [N, D] (scores2 dead after softmax)

  // 1. Convert inputs to bf16.
  cvt_f32_to_bf16<<<(int)(ND / 1024), 256, 0, stream>>>(x1, x1b, (long)ND);
  cvt_f32_to_bf16<<<(int)(DDn / 1024), 256, 0, stream>>>(x2, x2b, (long)DDn);
  cvt_f32_to_bf16<<<(int)(DDn / 1024), 256, 0, stream>>>(x3, x3b, (long)DDn);
  cvt_f32_to_bf16<<<(int)(DDn / 1024), 256, 0, stream>>>(W1, W1b, (long)DDn);
  cvt_f32_to_bf16<<<(int)(DDn / 1024), 256, 0, stream>>>(W2, W2b, (long)DDn);

  const dim3 blk(256);
  // 2. h1 = tanh(x1 @ W1^T + b1)                [N, D]
  gemm_wmma_bf16<true, true, true, true>
      <<<dim3(DD / BN, NN / BM), blk, 0, stream>>>(x1b, W1b, b1, nullptr, h1b, NN, DD, DD);
  // 3. p2 = x2 @ W2^T + b2                      [D, D]
  gemm_wmma_bf16<true, true, false, true>
      <<<dim3(DD / BN, DD / BM), blk, 0, stream>>>(x2b, W2b, b2, nullptr, p2b, DD, DD, DD);
  // 4. scores = h1 @ p2^T                       [N, D]
  gemm_wmma_bf16<true, true, false, false>
      <<<dim3(DD / BN, NN / BM), blk, 0, stream>>>(h1b, p2b, nullptr, nullptr, scb, NN, DD, DD);
  // 5. h3 = tanh(x3 @ W1^T + b1)                [D, D]
  gemm_wmma_bf16<true, true, true, true>
      <<<dim3(DD / BN, DD / BM), blk, 0, stream>>>(x3b, W1b, b1, nullptr, h3b, DD, DD, DD);
  // 6. scores2 = h3 @ scores^T                  [D, N] fp32
  gemm_wmma_bf16<true, false, false, false>
      <<<dim3(NN / BN, DD / BM), blk, 0, stream>>>(h3b, scb, nullptr, s2f, nullptr, DD, NN, DD);
  // 7. weights = softmax(scores2, axis=1)       [D, N] bf16
  softmax_rows_f32_to_bf16<<<DD, 256, 0, stream>>>(s2f, weights, NN);
  // 8. weightsT                                  [N, D]
  transpose_bf16<<<dim3(NN / 32, DD / 32), blk, 0, stream>>>(weights, weightsT, DD, NN);

  float* att1 = (float*)d_out;
  float* att2 = att1 + DDn;
  float* att3 = att2 + ND;
  // 9.  att1 = weights @ x1                     [D, D]
  gemm_wmma_bf16<false, false, false, false>
      <<<dim3(DD / BN, DD / BM), blk, 0, stream>>>(weights, x1b, nullptr, att1, nullptr, DD, DD, NN);
  // 10. att2 = weightsT @ x2                    [N, D]
  gemm_wmma_bf16<false, false, false, false>
      <<<dim3(DD / BN, NN / BM), blk, 0, stream>>>(weightsT, x2b, nullptr, att2, nullptr, NN, DD, DD);
  // 11. att3 = weightsT @ x3                    [N, D]
  gemm_wmma_bf16<false, false, false, false>
      <<<dim3(DD / BN, NN / BM), blk, 0, stream>>>(weightsT, x3b, nullptr, att3, nullptr, NN, DD, DD);
}
